// JointStreamAttention_14474039787743
// MI455X (gfx1250) — 8-bit matrix operand order repaired and checked by reading, compiled, not run on hardware
//
#include <hip/hip_runtime.h>
#include <stdint.h>

typedef __attribute__((ext_vector_type(8))) int v8i;

#define D_MODEL   2048
#define T_JOINT   4608
#define BATCH     4
#define S_SELF    4096
#define S_CROSS   512
#define ROWS_SELF (BATCH * S_SELF)
#define ROWS_CROSS (BATCH * S_CROSS)
#define ROWS_JOINT (BATCH * T_JOINT)
#define W_ELEMS   (2048ull * 2048ull)
#define EPSF      1e-5f
#define BIGSEG    (1 << 28)

static const size_t OFF_WSUM   = 0;
static const size_t OFF_WPART  = 4096;
static const size_t OFF_WDEQ   = 16384;
static const size_t OFF_ASCALE = 32768;
static const size_t OFF_ACT8   = 1ull << 20;
static const size_t OFF_W8     = 64ull << 20;
static const size_t OFF_BUF0   = 128ull << 20;
static const size_t BUF_ELEMS  = (size_t)ROWS_JOINT * D_MODEL;

struct WPtrs { const float* p[8]; };

__global__ __launch_bounds__(256)
void wabs_partial_kernel(WPtrs wp, float* __restrict__ part) {
  __shared__ float red[256];
  const float* W = wp.p[blockIdx.y];
  float s = 0.f;
  for (size_t i = (size_t)blockIdx.x * 256 + threadIdx.x; i < W_ELEMS; i += 256ull * 256ull)
    s += fabsf(W[i]);
  red[threadIdx.x] = s; __syncthreads();
  for (int o = 128; o > 0; o >>= 1) {
    if ((int)threadIdx.x < o) red[threadIdx.x] += red[threadIdx.x + o];
    __syncthreads();
  }
  if (threadIdx.x == 0) part[blockIdx.y * 256 + blockIdx.x] = red[0];
}

__global__ __launch_bounds__(256)
void wabs_final_kernel(const float* __restrict__ part, float* __restrict__ wsum) {
  __shared__ float red[256];
  red[threadIdx.x] = part[blockIdx.x * 256 + threadIdx.x]; __syncthreads();
  for (int o = 128; o > 0; o >>= 1) {
    if ((int)threadIdx.x < o) red[threadIdx.x] += red[threadIdx.x + o];
    __syncthreads();
  }
  if (threadIdx.x == 0) wsum[blockIdx.x] = red[0];
}

__global__ __launch_bounds__(256)
void weight_quant_kernel(WPtrs wp, const float* __restrict__ wsum,
                         int8_t* __restrict__ w8base, float* __restrict__ wdeq) {
  const int w = blockIdx.y;
  const float* W = wp.p[w];
  int8_t* Q = w8base + (size_t)w * W_ELEMS;
  const float mean  = wsum[w] * (1.f / (float)W_ELEMS);
  const float denom = fmaxf(mean, 1e-5f);
  const float scale = 1.f / denom;
  const size_t base = (size_t)blockIdx.x * 1024 + threadIdx.x;
#pragma unroll
  for (int i = 0; i < 4; ++i) {
    size_t idx = base + (size_t)i * 256;
    float q = rintf(W[idx] * scale);
    q = fminf(fmaxf(q, -1.f), 1.f);
    Q[idx] = (int8_t)q;
  }
  if (blockIdx.x == 0 && threadIdx.x == 0) wdeq[w] = denom;
}

__global__ __launch_bounds__(256)
void act_quant_kernel(const float* __restrict__ X, int8_t* __restrict__ Q,
                      float* __restrict__ rscale) {
  __shared__ float red[256];
  const size_t row = blockIdx.x;
  const float* x = X + row * (size_t)D_MODEL;
  float v[8]; float ss = 0.f;
#pragma unroll
  for (int i = 0; i < 8; ++i) { v[i] = x[threadIdx.x + i * 256]; ss += v[i] * v[i]; }
  red[threadIdx.x] = ss; __syncthreads();
  for (int o = 128; o > 0; o >>= 1) {
    if ((int)threadIdx.x < o) red[threadIdx.x] += red[threadIdx.x + o];
    __syncthreads();
  }
  const float rms = rsqrtf(red[0] * (1.f / (float)D_MODEL) + EPSF);
  __syncthreads();
  float ma = 0.f;
#pragma unroll
  for (int i = 0; i < 8; ++i) { v[i] *= rms; ma = fmaxf(ma, fabsf(v[i])); }
  red[threadIdx.x] = ma; __syncthreads();
  for (int o = 128; o > 0; o >>= 1) {
    if ((int)threadIdx.x < o) red[threadIdx.x] = fmaxf(red[threadIdx.x], red[threadIdx.x + o]);
    __syncthreads();
  }
  const float amax  = fmaxf(red[0], 1e-5f);
  const float scale = 127.f / amax;
  int8_t* q = Q + row * (size_t)D_MODEL;
#pragma unroll
  for (int i = 0; i < 8; ++i) {
    float t = rintf(v[i] * scale);
    t = fminf(fmaxf(t, -128.f), 127.f);
    q[threadIdx.x + i * 256] = (int8_t)t;
  }
  if (threadIdx.x == 0) rscale[row] = amax * (1.f / 127.f);
}

template <int IN_BASE, int IN_SEG, int IN_STRIDE,
          int OUT_BASE, int OUT_SEG, int OUT_STRIDE>
__global__ __launch_bounds__(256)
void gemm_i8_kernel(const int8_t* __restrict__ A8, const float* __restrict__ rscale,
                    const int8_t* __restrict__ W8, const float* __restrict__ wdeq, int widx,
                    float* __restrict__ C) {
  const int K = D_MODEL;
  const int wave = threadIdx.x >> 5;
  const int lane = threadIdx.x & 31;
  const int ml = lane & 15;
  const int hl = lane >> 4;
  const int wm = wave & 1;
  const int wn = wave >> 1;
  const int m0 = blockIdx.x * 32 + wm * 16;
  const int n0 = blockIdx.y * 128 + wn * 32;

  const long arow0 = (long)IN_BASE + (long)(m0 / IN_SEG) * IN_STRIDE + (m0 % IN_SEG);
  const long orow0 = (long)OUT_BASE + (long)(m0 / OUT_SEG) * OUT_STRIDE + (m0 % OUT_SEG);

  const int8_t* Arow = A8 + (arow0 + ml) * (long)K;
  const int8_t* B0   = W8 + (long)(n0 + ml) * K;
  const int8_t* B1   = W8 + (long)(n0 + 16 + ml) * K;

  v8i acc0 = {}, acc1 = {};
#pragma unroll 2
  for (int k0 = 0; k0 < K; k0 += 64) {
    const int8_t* ap = Arow + k0 + hl * 16;
    int2 a01 = *(const int2*)(ap);
    int2 a23 = *(const int2*)(ap + 8);
    int2 a45 = *(const int2*)(ap + 32);
    int2 a67 = *(const int2*)(ap + 40);
    const int8_t* bp0 = B0 + k0 + hl * 16;
    int4 p03 = *(const int4*)(bp0);
    int4 p47 = *(const int4*)(bp0 + 32);
    const int8_t* bp1 = B1 + k0 + hl * 16;
    int4 q03 = *(const int4*)(bp1);
    int4 q47 = *(const int4*)(bp1 + 32);

    v8i a, b, c;
    a[0] = a01.x; a[1] = a01.y; a[2] = a23.x; a[3] = a23.y;
    a[4] = a45.x; a[5] = a45.y; a[6] = a67.x; a[7] = a67.y;
    b[0] = p03.x; b[1] = p03.y; b[2] = p03.z; b[3] = p03.w;
    b[4] = p47.x; b[5] = p47.y; b[6] = p47.z; b[7] = p47.w;
    c[0] = q03.x; c[1] = q03.y; c[2] = q03.z; c[3] = q03.w;
    c[4] = q47.x; c[5] = q47.y; c[6] = q47.z; c[7] = q47.w;

    acc0 = __builtin_amdgcn_wmma_i32_16x16x64_iu8(true, a, true, b, acc0, false, false);
    acc1 = __builtin_amdgcn_wmma_i32_16x16x64_iu8(true, a, true, c, acc1, false, false);
  }

  const float wd = wdeq[widx];
#pragma unroll
  for (int r = 0; r < 8; ++r) {
    const int rr = r + hl * 8;
    const float s = rscale[arow0 + rr] * wd;
    float* crow = C + (orow0 + rr) * (long)D_MODEL;
    crow[n0 + ml]      = (float)acc0[r] * s;
    crow[n0 + 16 + ml] = (float)acc1[r] * s;
  }
}

__global__ __launch_bounds__(256)
void hgrn_scan_kernel(const float* __restrict__ Iraw, const float* __restrict__ Fraw,
                      float* __restrict__ O) {
  const int tid = blockIdx.x * blockDim.x + threadIdx.x;
  const int b = tid >> 11;
  const int d = tid & 2047;
  const long base = ((long)b * T_JOINT) * D_MODEL + d;
  float h = 0.f;
  for (int t = 0; t < T_JOINT; ++t) {
    const long idx = base + (long)t * D_MODEL;
    const float iv = Iraw[idx];
    const float fv = Fraw[idx];
    const float sig_f = 1.f / (1.f + expf(-fv));
    const float logf_ = fminf(fv, 0.f) - log1pf(expf(-fabsf(fv)));
    const float inp = (iv / (1.f + expf(-iv))) * (1.f - sig_f);
    h = expf(logf_) * h + inp;
    O[idx] = h;
  }
}

__global__ __launch_bounds__(256)
void gate_norm_kernel(const float* __restrict__ O, const float* __restrict__ G,
                      const float* __restrict__ gw, float* __restrict__ OUT) {
  const int wave = threadIdx.x >> 5;
  const int lane = threadIdx.x & 31;
  const size_t gh = (size_t)blockIdx.x * 8 + wave;
  const size_t base = gh * 128;
  float ov[4]; float ss = 0.f;
#pragma unroll
  for (int i = 0; i < 4; ++i) { ov[i] = O[base + lane + i * 32]; ss += ov[i] * ov[i]; }
#pragma unroll
  for (int off = 16; off > 0; off >>= 1) ss += __shfl_xor(ss, off, 32);
  const float rms = rsqrtf(ss * (1.f / 128.f) + EPSF);
#pragma unroll
  for (int i = 0; i < 4; ++i) {
    const int j = lane + i * 32;
    const float g  = G[base + j];
    const float sg = 1.f / (1.f + expf(-g));
    OUT[base + j] = ov[i] * rms * gw[j] * g * sg;
  }
}

extern "C" void kernel_launch(void* const* d_in, const int* in_sizes, int n_in,
                              void* d_out, int out_size, void* d_ws, size_t ws_size,
                              hipStream_t stream) {
  (void)in_sizes; (void)n_in; (void)out_size; (void)ws_size;

  const float* hidden = (const float*)d_in[0];
  const float* enc    = (const float*)d_in[1];
  const float* gnw    = (const float*)d_in[14];

  char* ws = (char*)d_ws;
  float*  wsum   = (float*)(ws + OFF_WSUM);
  float*  wpart  = (float*)(ws + OFF_WPART);
  float*  wdeq   = (float*)(ws + OFF_WDEQ);
  float*  ascale = (float*)(ws + OFF_ASCALE);
  int8_t* act8   = (int8_t*)(ws + OFF_ACT8);
  int8_t* w8     = (int8_t*)(ws + OFF_W8);
  float*  buf0   = (float*)(ws + OFF_BUF0);
  float*  buf1   = buf0 + BUF_ELEMS;
  float*  buf2   = buf1 + BUF_ELEMS;
  float*  buf3   = buf2 + BUF_ELEMS;
  float*  out_s  = (float*)d_out;
  float*  out_c  = out_s + (size_t)ROWS_SELF * D_MODEL;

  WPtrs wp;
  wp.p[0] = (const float*)d_in[2];
  wp.p[1] = (const float*)d_in[6];
  wp.p[2] = (const float*)d_in[10];
  wp.p[3] = (const float*)d_in[11];
  wp.p[4] = (const float*)d_in[12];
  wp.p[5] = (const float*)d_in[13];
  wp.p[6] = (const float*)d_in[5];
  wp.p[7] = (const float*)d_in[9];

  wabs_partial_kernel<<<dim3(256, 8), 256, 0, stream>>>(wp, wpart);
  wabs_final_kernel<<<8, 256, 0, stream>>>(wpart, wsum);
  weight_quant_kernel<<<dim3(4096, 8), 256, 0, stream>>>(wp, wsum, w8, wdeq);

  act_quant_kernel<<<ROWS_SELF, 256, 0, stream>>>(hidden, act8, ascale);
  act_quant_kernel<<<ROWS_CROSS, 256, 0, stream>>>(enc, act8 + (size_t)ROWS_SELF * D_MODEL,
                                                   ascale + ROWS_SELF);

  gemm_i8_kernel<0, BIGSEG, 0, S_CROSS, S_SELF, T_JOINT>
      <<<dim3(ROWS_SELF / 32, 16), 256, 0, stream>>>(
      act8, ascale, w8 + 0 * W_ELEMS, wdeq, 0, buf0);
  gemm_i8_kernel<ROWS_SELF, BIGSEG, 0, 0, S_CROSS, T_JOINT>
      <<<dim3(ROWS_CROSS / 32, 16), 256, 0, stream>>>(
      act8, ascale, w8 + 1 * W_ELEMS, wdeq, 1, buf0);

  act_quant_kernel<<<ROWS_JOINT, 256, 0, stream>>>(buf0, act8, ascale);
  gemm_i8_kernel<0, BIGSEG, 0, 0, BIGSEG, 0>
      <<<dim3(ROWS_JOINT / 32, 16), 256, 0, stream>>>(
      act8, ascale, w8 + 2 * W_ELEMS, wdeq, 2, buf1);
  gemm_i8_kernel<0, BIGSEG, 0, 0, BIGSEG, 0>
      <<<dim3(ROWS_JOINT / 32, 16), 256, 0, stream>>>(
      act8, ascale, w8 + 3 * W_ELEMS, wdeq, 3, buf2);
  gemm_i8_kernel<0, BIGSEG, 0, 0, BIGSEG, 0>
      <<<dim3(ROWS_JOINT / 32, 16), 256, 0, stream>>>(
      act8, ascale, w8 + 4 * W_ELEMS, wdeq, 4, buf3);

  hgrn_scan_kernel<<<(BATCH * D_MODEL) / 256, 256, 0, stream>>>(buf1, buf2, buf0);

  gate_norm_kernel<<<(BATCH * T_JOINT * 16) / 8, 256, 0, stream>>>(buf0, buf3, gnw, buf1);

  act_quant_kernel<<<ROWS_JOINT, 256, 0, stream>>>(buf1, act8, ascale);
  gemm_i8_kernel<0, BIGSEG, 0, 0, BIGSEG, 0>
      <<<dim3(ROWS_JOINT / 32, 16), 256, 0, stream>>>(
      act8, ascale, w8 + 5 * W_ELEMS, wdeq, 5, buf2);

  act_quant_kernel<<<ROWS_JOINT, 256, 0, stream>>>(buf2, act8, ascale);
  gemm_i8_kernel<S_CROSS, S_SELF, T_JOINT, 0, BIGSEG, 0>
      <<<dim3(ROWS_SELF / 32, 16), 256, 0, stream>>>(
      act8, ascale, w8 + 6 * W_ELEMS, wdeq, 6, out_s);
  gemm_i8_kernel<0, S_CROSS, T_JOINT, 0, BIGSEG, 0>
      <<<dim3(ROWS_CROSS / 32, 16), 256, 0, stream>>>(
      act8, ascale, w8 + 7 * W_ELEMS, wdeq, 7, out_c);
}
